// Twins_Spike_67336497266823
// MI455X (gfx1250) — matrix operand order repaired and checked by reading, not compiled, not run on hardware
//
#include <hip/hip_runtime.h>

typedef _Float16 half_t;
typedef __attribute__((ext_vector_type(16))) _Float16 v16h;
typedef __attribute__((ext_vector_type(8)))  _Float16 v8h;
typedef __attribute__((ext_vector_type(8)))  float    v8f;

#define BN_EPS 1e-5f

constexpr int NHEADS = 12;
constexpr int HD     = 64;
constexpr int CDIM   = 768;
constexpr int NTOK   = 1024;
constexpr int BATCH  = 8;
constexpr int ROWS   = BATCH * NTOK;   // 8192
constexpr int QKVC   = 3 * CDIM;       // 2304

// ---------------------------------------------------------------------------
// WMMA f16 GEMM: C[MxN] = A[MxK] @ B[KxN] (+bias), batched via grid.z.
// B is ALWAYS supplied pre-transposed (Bt = N x K row-major) so both operand
// tiles stage into LDS with 16B vector copies (no scalar LDS transpose).
// Register-staged double buffering: next K-tile global loads are issued right
// after the barrier and their latency is hidden behind the WMMA burst.
// Fragment layouts per CDNA5 ISA 7.12.2 (wave32):
//   A frag lane L, half j : A[m0 + (L&15), (j&7) + (j>=8?16:0) + (L>=16?8:0)]
//   B frag lane L, half j : Bt[n0 + (L&15), (j&7) + (j>=8?16:0) + (L>=16?8:0)]
//   C/D  lane L, vgpr r   : C[m0 + r + (L>=16?8:0), n0 + (L&15)]
// TRANSC: write C transposed (C^T is N x K... i.e. N-major), which makes each
// lane's 8 accumulators contiguous -> single v8h store (used for k^T v).
// ---------------------------------------------------------------------------
template<int BM, int BN, int WROWS, int WCOLS, bool OUTF16, bool BIAS, bool TRANSC>
__global__ __launch_bounds__(256) void gemm_wmma_f16(
    const half_t* __restrict__ A, long batchA,
    const half_t* __restrict__ Bt, long batchB,
    void* __restrict__ C, long batchC,
    const float* __restrict__ bias,
    int K, int lda, int ldb, int ldc)
{
    constexpr int BK = 32;
    constexpr int LSTR = BK + 8;                 // 40 halves (80B) pad
    constexpr int WTM = BM / (16 * WROWS);
    constexpr int WTN = BN / (16 * WCOLS);
    constexpr int ASEG = BM / 64;                // v8h segments per thread
    constexpr int BSEG = BN / 64;
    static_assert(WTM >= 1 && WTN >= 1, "bad wave tiling");

    __shared__ half_t As[BM * LSTR];
    __shared__ half_t Bs[BN * LSTR];

    const int tid  = threadIdx.x;
    const int wave = tid >> 5;
    const int lane = tid & 31;
    const int l15  = lane & 15;
    const int lhi  = lane >> 4;
    const int wm   = wave % WROWS;
    const int wn   = wave / WROWS;

    const long bz = blockIdx.z;
    const half_t* Ab = A  + bz * batchA + (long)blockIdx.y * BM * lda;
    const half_t* Bb = Bt + bz * batchB + (long)blockIdx.x * BN * ldb;

    // fixed staging coordinates for this thread
    int arow[ASEG], acol[ASEG], brow[BSEG], bcol[BSEG];
#pragma unroll
    for (int it = 0; it < ASEG; ++it) {
        int seg = it * 256 + tid;
        arow[it] = seg >> 2; acol[it] = (seg & 3) * 8;
    }
#pragma unroll
    for (int it = 0; it < BSEG; ++it) {
        int seg = it * 256 + tid;
        brow[it] = seg >> 2; bcol[it] = (seg & 3) * 8;
    }

    // prologue: stage tile 0 into registers
    v8h aReg[ASEG], bReg[BSEG];
#pragma unroll
    for (int it = 0; it < ASEG; ++it)
        aReg[it] = *(const v8h*)(Ab + (long)arow[it] * lda + acol[it]);
#pragma unroll
    for (int it = 0; it < BSEG; ++it)
        bReg[it] = *(const v8h*)(Bb + (long)brow[it] * ldb + bcol[it]);

    const int ktiles = K / BK;
    v8f acc[WTM][WTN] = {};

    for (int kt = 0; kt < ktiles; ++kt) {
        // staged regs -> LDS
#pragma unroll
        for (int it = 0; it < ASEG; ++it)
            *(v8h*)&As[arow[it] * LSTR + acol[it]] = aReg[it];
#pragma unroll
        for (int it = 0; it < BSEG; ++it)
            *(v8h*)&Bs[brow[it] * LSTR + bcol[it]] = bReg[it];
        __syncthreads();

        // issue next tile's global loads (latency hidden by WMMAs below)
        if (kt + 1 < ktiles) {
            const int k0 = (kt + 1) * BK;
#pragma unroll
            for (int it = 0; it < ASEG; ++it)
                aReg[it] = *(const v8h*)(Ab + (long)arow[it] * lda + k0 + acol[it]);
#pragma unroll
            for (int it = 0; it < BSEG; ++it)
                bReg[it] = *(const v8h*)(Bb + (long)brow[it] * ldb + k0 + bcol[it]);
        }

        // fragments from LDS + WMMA
        v16h afrag[WTM];
#pragma unroll
        for (int tm = 0; tm < WTM; ++tm) {
            const int row = (wm * WTM + tm) * 16 + l15;
            const half_t* p = &As[row * LSTR + lhi * 8];
            v8h lo = *(const v8h*)p;
            v8h hi = *(const v8h*)(p + 16);
            v16h a;
#pragma unroll
            for (int j = 0; j < 8; ++j) { a[j] = lo[j]; a[j + 8] = hi[j]; }
            afrag[tm] = a;
        }
#pragma unroll
        for (int tn = 0; tn < WTN; ++tn) {
            const int col = (wn * WTN + tn) * 16 + l15;
            const half_t* p = &Bs[col * LSTR + lhi * 8];
            v8h lo = *(const v8h*)p;
            v8h hi = *(const v8h*)(p + 16);
            v16h b;
#pragma unroll
            for (int j = 0; j < 8; ++j) { b[j] = lo[j]; b[j + 8] = hi[j]; }
#pragma unroll
            for (int tm = 0; tm < WTM; ++tm)
                acc[tm][tn] = __builtin_amdgcn_wmma_f32_16x16x32_f16(
                    false, afrag[tm], false, b, (short)0, acc[tm][tn], false, false);
        }
        __syncthreads();
    }

    // ---- writeback
#pragma unroll
    for (int tm = 0; tm < WTM; ++tm) {
#pragma unroll
        for (int tn = 0; tn < WTN; ++tn) {
            const int gn  = blockIdx.x * BN + (wn * WTN + tn) * 16 + l15;
            const long gm = (long)blockIdx.y * BM + (wm * WTM + tm) * 16 + lhi * 8;
            const float ba = BIAS ? bias[gn] : 0.0f;
            if (TRANSC && OUTF16) {
                // each lane's 8 values are contiguous along M in C^T -> one v8h
                v8h o;
#pragma unroll
                for (int r = 0; r < 8; ++r) o[r] = (half_t)(acc[tm][tn][r] + ba);
                *(v8h*)&((half_t*)C)[bz * batchC + (long)gn * ldc + gm] = o;
            } else {
#pragma unroll
                for (int r = 0; r < 8; ++r) {
                    float v = acc[tm][tn][r] + ba;
                    if (OUTF16)
                        ((half_t*)C)[bz * batchC + (gm + r) * ldc + gn] = (half_t)v;
                    else
                        ((float*)C)[bz * batchC + (gm + r) * ldc + gn] = v;
                }
            }
        }
    }
}

// ---------------------------------------------------------------------------
// Elementwise / reduction kernels
// ---------------------------------------------------------------------------
__global__ void zero_kernel(float* p, int n) {
    int i = blockIdx.x * blockDim.x + threadIdx.x;
    if (i < n) p[i] = 0.f;
}

// in: K x N row-major f32;  out: N x K row-major f16 (pre-transposed weights)
__global__ void cvt_transpose_f32_f16(const float* __restrict__ in,
                                      half_t* __restrict__ out,
                                      int Kdim, int Ndim)
{
    long i = (long)blockIdx.x * blockDim.x + threadIdx.x;
    if (i < (long)Kdim * Ndim) {
        int n = (int)(i % Ndim), k = (int)(i / Ndim);
        out[(long)n * Kdim + k] = (half_t)in[i];
    }
}

// stage 0: per-head BN, /thr, clamp[0,1]; accumulate hoyer sums per head.
__global__ __launch_bounds__(256) void stage0_bn_clamp(
    const float* __restrict__ x, const float* __restrict__ gamma,
    const float* __restrict__ beta, const float* __restrict__ mean,
    const float* __restrict__ var, const float* __restrict__ thr,
    half_t* __restrict__ xc, float* __restrict__ accum)
{
    __shared__ float s[NHEADS * 2];
    int tid = threadIdx.x;
    if (tid < NHEADS * 2) s[tid] = 0.f;
    __syncthreads();
    const long row = blockIdx.x;
    const float t = thr[0];
    for (int i = tid; i < CDIM; i += 256) {
        int h = i >> 6;
        float sc = gamma[h] * rsqrtf(var[h] + BN_EPS);
        float u = ((x[row * CDIM + i] - mean[h]) * sc + beta[h]) / t;
        u = fminf(fmaxf(u, 0.f), 1.f);
        xc[row * CDIM + i] = (half_t)u;
        atomicAdd(&s[h * 2 + 0], u * u);
        atomicAdd(&s[h * 2 + 1], u);
    }
    __syncthreads();
    if (tid < NHEADS * 2) atomicAdd(&accum[tid], s[tid]);
}

// split qkv rows: q -> (B,H,N,hd) f16 copy; k,v -> BN/clamp, stored TRANSPOSED
// as (B,H,hd,N) so the k^T v GEMM reads both operands contiguously.
__global__ __launch_bounds__(256) void split_qkv_bn(
    const half_t* __restrict__ qkv,
    const float* __restrict__ gamma, const float* __restrict__ beta,
    const float* __restrict__ mean, const float* __restrict__ var,
    const float* __restrict__ thr,
    half_t* __restrict__ q16, half_t* __restrict__ kT, half_t* __restrict__ vT,
    float* __restrict__ accum)
{
    __shared__ float s[2 * NHEADS * 2];
    int tid = threadIdx.x;
    if (tid < 2 * NHEADS * 2) s[tid] = 0.f;
    __syncthreads();
    const long row = blockIdx.x;
    const int b = (int)(row >> 10), n = (int)(row & 1023);
    for (int i = tid; i < QKVC; i += 256) {
        int which = i / CDIM;
        int rem = i - which * CDIM;
        int h = rem >> 6, d = rem & 63;
        float val = (float)qkv[row * QKVC + i];
        if (which == 0) {
            q16[((long)(b * NHEADS + h) * NTOK + n) * HD + d] = (half_t)val;
        } else {
            int p = which * NHEADS + h;
            float sc = gamma[p] * rsqrtf(var[p] + BN_EPS);
            float u = ((val - mean[p]) * sc + beta[p]) / thr[which];
            u = fminf(fmaxf(u, 0.f), 1.f);
            long o = ((long)(b * NHEADS + h) * HD + d) * NTOK + n;
            if (which == 1) kT[o] = (half_t)u; else vT[o] = (half_t)u;
            int sl = (which - 1) * (NHEADS * 2) + h * 2;
            atomicAdd(&s[sl], u * u);
            atomicAdd(&s[sl + 1], u);
        }
    }
    __syncthreads();
    if (tid < 2 * NHEADS * 2) atomicAdd(&accum[NHEADS * 2 + tid], s[tid]);
}

__global__ void hoyer_thresholds(const float* __restrict__ accum,
                                 float* __restrict__ hthr, int s0, int cnt)
{
    int i = threadIdx.x;
    if (i < cnt * NHEADS) {
        int st = s0 + i / NHEADS, h = i % NHEADS;
        hthr[st * NHEADS + h] = accum[st * NHEADS * 2 + h * 2] /
                                (accum[st * NHEADS * 2 + h * 2 + 1] + 1e-9f);
    }
}

__global__ void binarize_x(const half_t* __restrict__ xc,
                           const float* __restrict__ hthr0,
                           half_t* __restrict__ x2)
{
    long i = (long)blockIdx.x * blockDim.x + threadIdx.x;
    int h = (int)((i % CDIM) >> 6);
    x2[i] = ((float)xc[i] >= hthr0[h]) ? (half_t)1.f : (half_t)0.f;
}

__global__ void binarize_kv(half_t* __restrict__ kT, half_t* __restrict__ vT,
                            const float* __restrict__ hthr)
{
    long i = (long)blockIdx.x * blockDim.x + threadIdx.x;
    int h = (int)((i >> 16) % NHEADS);      // 64*1024 elems per (b,h)
    kT[i] = ((float)kT[i] >= hthr[NHEADS + h])     ? (half_t)1.f : (half_t)0.f;
    vT[i] = ((float)vT[i] >= hthr[2 * NHEADS + h]) ? (half_t)1.f : (half_t)0.f;
}

// stage 3: BN/clamp over y; the (B,H,N,hd)->(B,H,hd,N) raw view keeps the
// head channel identical, so per-head BN + hoyer are layout-invariant here.
__global__ __launch_bounds__(256) void stage3_bn_clamp(
    const half_t* __restrict__ y, const float* __restrict__ gamma,
    const float* __restrict__ beta, const float* __restrict__ mean,
    const float* __restrict__ var, const float* __restrict__ thr,
    half_t* __restrict__ yc, float* __restrict__ accum)
{
    __shared__ float s[2];
    int tid = threadIdx.x;
    if (tid < 2) s[tid] = 0.f;
    __syncthreads();
    const int h = (blockIdx.x >> 4) % NHEADS;     // 16 blocks per (b,h)
    const long base = (long)blockIdx.x * 4096;
    const float sc = gamma[h] * rsqrtf(var[h] + BN_EPS);
    const float mu = mean[h], bb = beta[h], t = thr[0];
    float pn = 0.f, pd = 0.f;
    for (int it = 0; it < 16; ++it) {
        long idx = base + it * 256 + tid;
        float u = (((float)y[idx] - mu) * sc + bb) / t;
        u = fminf(fmaxf(u, 0.f), 1.f);
        yc[idx] = (half_t)u;
        pn += u * u; pd += u;
    }
    atomicAdd(&s[0], pn); atomicAdd(&s[1], pd);
    __syncthreads();
    if (tid == 0) {
        atomicAdd(&accum[3 * NHEADS * 2 + h * 2 + 0], s[0]);
        atomicAdd(&accum[3 * NHEADS * 2 + h * 2 + 1], s[1]);
    }
}

// binarize stage-3 and scatter into z[(b,n2), h*64+d2] per the reshape view.
__global__ void binarize_scatter_y(const half_t* __restrict__ yc,
                                   const float* __restrict__ hthr3,
                                   half_t* __restrict__ z)
{
    long i = (long)blockIdx.x * blockDim.x + threadIdx.x;
    int b = (int)(i / ((long)NHEADS * 65536));
    long r = i % ((long)NHEADS * 65536);
    int h = (int)(r >> 16);
    int flat = (int)(r & 65535);
    int d2 = flat >> 10, n2 = flat & 1023;
    half_t sp = ((float)yc[i] >= hthr3[h]) ? (half_t)1.f : (half_t)0.f;
    z[((long)(b * NTOK + n2)) * CDIM + h * HD + d2] = sp;
}

// ---------------------------------------------------------------------------
extern "C" void kernel_launch(void* const* d_in, const int* in_sizes, int n_in,
                              void* d_out, int out_size, void* d_ws, size_t ws_size,
                              hipStream_t stream)
{
    (void)in_sizes; (void)n_in; (void)out_size; (void)ws_size;
    const float* x      = (const float*)d_in[0];
    const float* gamma  = (const float*)d_in[1];   // (4,12)
    const float* beta   = (const float*)d_in[2];
    const float* mean   = (const float*)d_in[3];
    const float* var    = (const float*)d_in[4];
    const float* thr    = (const float*)d_in[5];   // (4,)
    const float* w_qkv  = (const float*)d_in[6];   // (768,2304)
    const float* w_proj = (const float*)d_in[7];   // (768,768)
    const float* b_proj = (const float*)d_in[8];   // (768,)
    float* out = (float*)d_out;

    char* ws = (char*)d_ws;
    size_t off = 0;
    auto take = [&](size_t bytes) -> char* {
        char* p = ws + off;
        off = (off + bytes + 255) & ~(size_t)255;
        return p;
    };

    float*  accum    = (float*) take(4 * NHEADS * 2 * sizeof(float));
    float*  hthr     = (float*) take(4 * NHEADS * sizeof(float));
    half_t* wqkvT16  = (half_t*)take((size_t)CDIM * QKVC * sizeof(half_t)); // 2304x768
    half_t* wprojT16 = (half_t*)take((size_t)CDIM * CDIM * sizeof(half_t)); // 768x768
    half_t* xc       = (half_t*)take((size_t)ROWS * CDIM * sizeof(half_t));
    half_t* x2       = (half_t*)take((size_t)ROWS * CDIM * sizeof(half_t));
    half_t* qkv16    = (half_t*)take((size_t)ROWS * QKVC * sizeof(half_t));
    half_t* q16      = (half_t*)take((size_t)ROWS * CDIM * sizeof(half_t));
    half_t* kT       = (half_t*)take((size_t)ROWS * CDIM * sizeof(half_t));
    half_t* vT       = (half_t*)take((size_t)ROWS * CDIM * sizeof(half_t));
    half_t* ktvT16   = (half_t*)take((size_t)BATCH * NHEADS * HD * HD * sizeof(half_t));
    half_t* ybuf     = (half_t*)take((size_t)ROWS * CDIM * sizeof(half_t));
    half_t* yc = xc;   // xc dead after binarize_x -> reuse
    half_t* z  = x2;   // x2 dead after qkv GEMM  -> reuse

    // 0) zero hoyer accumulators (ws is poisoned; must re-zero every call)
    zero_kernel<<<1, 128, 0, stream>>>(accum, 4 * NHEADS * 2);
    // 1) weights -> f16, pre-transposed to N x K so all GEMM B-tiles are
    //    straight 16B vector copies into LDS
    cvt_transpose_f32_f16<<<(CDIM * QKVC) / 256, 256, 0, stream>>>(
        w_qkv, wqkvT16, CDIM, QKVC);
    cvt_transpose_f32_f16<<<(CDIM * CDIM) / 256, 256, 0, stream>>>(
        w_proj, wprojT16, CDIM, CDIM);
    // 2) stage0 BN/clamp + hoyer -> binary x2
    stage0_bn_clamp<<<ROWS, 256, 0, stream>>>(x, gamma, beta, mean, var, thr, xc, accum);
    hoyer_thresholds<<<1, 64, 0, stream>>>(accum, hthr, 0, 1);
    binarize_x<<<(ROWS * CDIM) / 256, 256, 0, stream>>>(xc, hthr, x2);
    // 3) qkv = x2 @ w_qkv : 8192x2304x768, f16 out
    gemm_wmma_f16<128, 128, 4, 2, true, false, false>
        <<<dim3(QKVC / 128, ROWS / 128, 1), 256, 0, stream>>>(
        x2, 0, wqkvT16, 0, qkv16, 0, nullptr, CDIM, CDIM, CDIM, QKVC);
    // 4) split q/k/v, BN/clamp+hoyer on k,v -> binary kT,vT (B,H,hd,N)
    split_qkv_bn<<<ROWS, 256, 0, stream>>>(qkv16, gamma, beta, mean, var, thr,
                                           q16, kT, vT, accum);
    hoyer_thresholds<<<1, 64, 0, stream>>>(accum, hthr, 1, 2);
    binarize_kv<<<(ROWS * CDIM) / 256, 256, 0, stream>>>(kT, vT, hthr);
    // 5) ktv^T per (b,h): 64x64x1024; A=kT, Bt=vT; TRANSC -> vectorized stores,
    //    and output is directly the pre-transposed B for step 6.
    gemm_wmma_f16<64, 64, 4, 2, true, false, true>
        <<<dim3(1, 1, BATCH * NHEADS), 256, 0, stream>>>(
        kT, (long)HD * NTOK, vT, (long)HD * NTOK, ktvT16, (long)HD * HD, nullptr,
        NTOK, NTOK, NTOK, HD);
    // 6) y = q @ ktv per (b,h): 1024x64x64 (softmax-free attention, associativity)
    gemm_wmma_f16<128, 64, 8, 1, true, false, false>
        <<<dim3(1, NTOK / 128, BATCH * NHEADS), 256, 0, stream>>>(
        q16, (long)NTOK * HD, ktvT16, (long)HD * HD, ybuf, (long)NTOK * HD, nullptr,
        HD, HD, HD, HD);
    // 7) stage3 BN/clamp + hoyer, binarize + scatter into (B,N,C) layout
    stage3_bn_clamp<<<BATCH * NHEADS * 16, 256, 0, stream>>>(
        ybuf, gamma + 3 * NHEADS, beta + 3 * NHEADS, mean + 3 * NHEADS,
        var + 3 * NHEADS, thr + 3, yc, accum);
    hoyer_thresholds<<<1, 64, 0, stream>>>(accum, hthr, 3, 1);
    binarize_scatter_y<<<(ROWS * CDIM) / 256, 256, 0, stream>>>(yc, hthr + 3 * NHEADS, z);
    // 8) out = z @ w_proj + b_proj : 8192x768x768, f32 out
    gemm_wmma_f16<128, 128, 4, 2, false, true, false>
        <<<dim3(CDIM / 128, ROWS / 128, 1), 256, 0, stream>>>(
        z, 0, wprojT16, 0, out, 0, b_proj, CDIM, CDIM, CDIM, CDIM);
}
